// EncoderLayer_317827579927
// MI455X (gfx1250) — matrix operand order repaired and checked by reading, not compiled, not run on hardware
//
#include <hip/hip_runtime.h>
#include <hip/hip_bf16.h>
#include <math.h>

// ---------------- types ----------------
typedef _Float16 h16;
typedef __attribute__((ext_vector_type(8)))  _Float16 v8h;
typedef __attribute__((ext_vector_type(16))) _Float16 v16h;
typedef __attribute__((ext_vector_type(8)))  float    v8f;
typedef __attribute__((ext_vector_type(4)))  unsigned int v4u;
typedef __attribute__((ext_vector_type(4)))  int v4i;
typedef __attribute__((ext_vector_type(8)))  int v8i;

union V16 { v16h v; v8h h[2]; };

#define H_DIM 1024
#define NHEAD 16
#define DKH   64
#define FFN_D 4096
#define SEQ   1024
#define BATCH 8
#define ROWS  (BATCH * SEQ)   // 8192

#define HAS_TDM __has_builtin(__builtin_amdgcn_tensor_load_to_lds)

__device__ __forceinline__ v8h ld8(const h16* p) { return *(const v8h*)p; }

#if HAS_TDM
// Issue one TDM descriptor (D# groups 0 and 1; groups 2/3 zero for 2D tiles).
__device__ __forceinline__ void tdm_load(v4u g0, v8i g1) {
    v4i z4 = {0, 0, 0, 0};
#if __clang_major__ >= 23
    v8i z8 = {0, 0, 0, 0, 0, 0, 0, 0};
    __builtin_amdgcn_tensor_load_to_lds(g0, g1, z4, z4, z8, 0);
#else
    __builtin_amdgcn_tensor_load_to_lds(g0, g1, z4, z4, 0);
#endif
}
#endif

// ---------------- weight convert + transpose: W[K][N] f32 -> Wt[N][K] f16 ----
__global__ __launch_bounds__(256)
void cvt_transpose_kernel(const float* __restrict__ W, h16* __restrict__ Wt,
                          int K, int N) {
    size_t i = (size_t)blockIdx.x * 256 + threadIdx.x;
    size_t total = (size_t)K * N;
    if (i >= total) return;
    int n = (int)(i / K);
    int k = (int)(i % K);
    Wt[i] = (h16)W[(size_t)k * N + n];
}

// ---------------- LayerNorm (fp32 in, f16 out), one block per row ----------
__global__ __launch_bounds__(256)
void ln_f16_kernel(const float* __restrict__ x, const float* __restrict__ g,
                   const float* __restrict__ b, h16* __restrict__ y) {
    __shared__ float red[256];
    const int row = blockIdx.x;
    const int t = threadIdx.x;
    const float* xr = x + (size_t)row * H_DIM;
    float v[4];
    float s = 0.f;
#pragma unroll
    for (int i = 0; i < 4; ++i) { v[i] = xr[t + i * 256]; s += v[i]; }
    red[t] = s; __syncthreads();
    for (int o = 128; o; o >>= 1) { if (t < o) red[t] += red[t + o]; __syncthreads(); }
    const float mu = red[0] * (1.0f / H_DIM);
    __syncthreads();
    s = 0.f;
#pragma unroll
    for (int i = 0; i < 4; ++i) { float d = v[i] - mu; s += d * d; }
    red[t] = s; __syncthreads();
    for (int o = 128; o; o >>= 1) { if (t < o) red[t] += red[t + o]; __syncthreads(); }
    const float rstd = rsqrtf(red[0] * (1.0f / H_DIM) + 1e-5f);
    h16* yr = y + (size_t)row * H_DIM;
#pragma unroll
    for (int i = 0; i < 4; ++i) {
        int c = t + i * 256;
        yr[c] = (h16)((v[i] - mu) * rstd * g[c] + b[c]);
    }
}

// ---------------- V transpose: v[b,s,h*64+d] -> vt[(b*NH+h)*64+d][s] -------
__global__ __launch_bounds__(256)
void transpose_v_kernel(const h16* __restrict__ v, h16* __restrict__ vt) {
    size_t i = (size_t)blockIdx.x * 256 + threadIdx.x;
    if (i >= (size_t)ROWS * H_DIM) return;
    int bs = (int)(i / H_DIM);        // b*SEQ + s
    int c  = (int)(i % H_DIM);        // h*DKH + d
    int b  = bs / SEQ, s = bs % SEQ;
    int h  = c / DKH,  d = c % DKH;
    vt[((size_t)((b * NHEAD + h) * DKH + d)) * SEQ + s] = v[i];
}

// ---------------- WMMA GEMM: C = A(MxK,f16) * Bt(NxK,f16)^T + bias ---------
// Block tile 128x128, BK=64, 8 waves (2x4), each wave 64x32 = 4x2 WMMA tiles.
// Tile staging via Tensor Data Mover (pad feature produces LDS stride 72).
template <bool GELU, bool RESID, bool F32OUT>
__global__ __launch_bounds__(256)
void gemm_f16_kernel(const h16* __restrict__ A, const h16* __restrict__ Bt,
                     const float* __restrict__ bias, const float* __restrict__ resid,
                     float* __restrict__ out32, h16* __restrict__ out16,
                     int M, int N, int K) {
    constexpr int BK = 64;
    constexpr int LS = 72;  // 64 data halves + 8 pad halves per row (144B, 16B-aligned)
    __shared__ __align__(16) h16 As[128 * LS];
    __shared__ __align__(16) h16 Bs[128 * LS];

    const int t    = threadIdx.x;
    const int lane = t & 31;
    const int wave = t >> 5;
    const int wm   = wave >> 2;   // 0..1
    const int wn   = wave & 3;    // 0..3
    const int l16  = lane & 15;
    const int lh   = lane >> 4;   // 0/1

    const int m0 = blockIdx.y * 128;
    const int n0 = blockIdx.x * 128;

    v8f acc[4][2];
    const v8f zf = {0.f, 0.f, 0.f, 0.f, 0.f, 0.f, 0.f, 0.f};
#pragma unroll
    for (int i = 0; i < 4; ++i)
#pragma unroll
        for (int j = 0; j < 2; ++j) acc[i][j] = zf;

#if HAS_TDM
    // D# group1: data_size=2B, pad_enable, pad after 32 dwords (128B) by 4 dwords (16B),
    // tensor_dim0 = K, tensor_dim1 = 128, tile = 64 x 128, dim0 stride = K.
    v8i g1;
    g1[0] = (int)((1u << 16) | (1u << 20) | (4u << 22) | (3u << 25));
    g1[1] = (int)(((unsigned)K & 0xFFFFu) << 16);
    g1[2] = (int)(((unsigned)K >> 16) | (128u << 16));
    g1[3] = (int)(64u << 16);
    g1[4] = 128;
    g1[5] = K;
    g1[6] = 0;
    g1[7] = 0;
    const unsigned ldsA = (unsigned)(size_t)&As[0];
    const unsigned ldsB = (unsigned)(size_t)&Bs[0];
    const unsigned long long gA =
        (unsigned long long)(size_t)(A + (size_t)m0 * K);
    const unsigned long long gB =
        (unsigned long long)(size_t)(Bt + (size_t)n0 * K);
#else
    const int lr = t >> 1;            // 0..127
    const int lc = (t & 1) * 32;      // 0 or 32
    const h16* Ag = A  + (size_t)(m0 + lr) * K + lc;
    const h16* Bg = Bt + (size_t)(n0 + lr) * K + lc;
#endif

    for (int kb = 0; kb < K; kb += BK) {
#if HAS_TDM
        if (t == 0) {
            const unsigned long long ga = gA + (unsigned long long)kb * 2;
            const unsigned long long gb = gB + (unsigned long long)kb * 2;
            v4u g0a = {1u, ldsA, (unsigned)ga,
                       (unsigned)((ga >> 32) & 0x1FFFFFFu) | 0x80000000u};
            v4u g0b = {1u, ldsB, (unsigned)gb,
                       (unsigned)((gb >> 32) & 0x1FFFFFFu) | 0x80000000u};
            tdm_load(g0a, g1);
            tdm_load(g0b, g1);
        }
        __builtin_amdgcn_s_wait_tensorcnt(0);
#else
#pragma unroll
        for (int j = 0; j < 4; ++j) {
            *(v8h*)&As[lr * LS + lc + j * 8] = ld8(Ag + kb + j * 8);
            *(v8h*)&Bs[lr * LS + lc + j * 8] = ld8(Bg + kb + j * 8);
        }
#endif
        __syncthreads();

#pragma unroll
        for (int ks = 0; ks < 2; ++ks) {
            V16 af[4], bf[2];
            const int kA = ks * 32 + lh * 8;   // A frag: lanes<16 K{0..7,16..23}
            const int kB = ks * 32 + lh * 8;   // B frag: lanes<16 K{0..7,16..23}
#pragma unroll
            for (int mt = 0; mt < 4; ++mt) {
                int row = wm * 64 + mt * 16 + l16;
                af[mt].h[0] = *(const v8h*)&As[row * LS + kA];
                af[mt].h[1] = *(const v8h*)&As[row * LS + kA + 16];
            }
#pragma unroll
            for (int nt = 0; nt < 2; ++nt) {
                int col = wn * 32 + nt * 16 + l16;
                bf[nt].h[0] = *(const v8h*)&Bs[col * LS + kB];
                bf[nt].h[1] = *(const v8h*)&Bs[col * LS + kB + 16];
            }
#pragma unroll
            for (int mt = 0; mt < 4; ++mt)
#pragma unroll
                for (int nt = 0; nt < 2; ++nt)
                    acc[mt][nt] = __builtin_amdgcn_wmma_f32_16x16x32_f16(
                        false, af[mt].v, false, bf[nt].v, (short)0, acc[mt][nt],
                        false, false);
        }
        __syncthreads();
    }

    // epilogue: C layout = VGPR r, lane l -> row = r + 8*(l>=16), col = l%16
#pragma unroll
    for (int mt = 0; mt < 4; ++mt) {
#pragma unroll
        for (int nt = 0; nt < 2; ++nt) {
            const int gr0 = m0 + wm * 64 + mt * 16 + 8 * lh;
            const int gc  = n0 + wn * 32 + nt * 16 + l16;
            const float bv = bias[gc];
#pragma unroll
            for (int r = 0; r < 8; ++r) {
                const int gr = gr0 + r;
                float val = acc[mt][nt][r] + bv;
                if (GELU) val = 0.5f * val * (1.0f + erff(val * 0.70710678118654752f));
                if (RESID) val += resid[(size_t)gr * N + gc];
                if (F32OUT) out32[(size_t)gr * N + gc] = val;
                else        out16[(size_t)gr * N + gc] = (h16)val;
            }
        }
    }
}

// ---------------- Flash attention: one wave per (b, h, 16-row q tile) ------
__global__ __launch_bounds__(32)
void attn_kernel(const h16* __restrict__ q, const h16* __restrict__ k,
                 const h16* __restrict__ vt, const float* __restrict__ bias,
                 const int* __restrict__ mask, h16* __restrict__ o) {
    __shared__ __align__(16) h16 Ps[16 * 48];

    const int lane = threadIdx.x;
    const int l16  = lane & 15;
    const int lh   = lane >> 4;
    const int qt   = blockIdx.x;      // 0..63
    const int h    = blockIdx.y;
    const int b    = blockIdx.z;
    const float scale = 0.125f;       // DK^-0.5 = 1/8

    // Q fragments (A-matrix 16x32 each), scale folded in
    V16 aQ[2];
    {
        const int qrow = qt * 16 + l16;
        const h16* qp = q + (size_t)(b * SEQ + qrow) * H_DIM + h * DKH;
#pragma unroll
        for (int f = 0; f < 2; ++f) {
            int ko = f * 32 + lh * 8;
            v8h lo = ld8(qp + ko);
            v8h hi = ld8(qp + ko + 16);
#pragma unroll
            for (int i = 0; i < 8; ++i) {
                lo[i] = (h16)((float)lo[i] * scale);
                hi[i] = (h16)((float)hi[i] * scale);
            }
            aQ[f].h[0] = lo;
            aQ[f].h[1] = hi;
        }
    }

    const v8f zf = {0.f, 0.f, 0.f, 0.f, 0.f, 0.f, 0.f, 0.f};
    float mrow[8], srow[8], corr[8];
    v8f oacc[4];
#pragma unroll
    for (int r = 0; r < 8; ++r) { mrow[r] = -1e30f; srow[r] = 0.f; }
#pragma unroll
    for (int tdk = 0; tdk < 4; ++tdk) oacc[tdk] = zf;

    const int row0 = qt * 16 + 8 * lh;
    const size_t bias_base = (size_t)((b * NHEAD + h) * SEQ) * SEQ;

    for (int kt = 0; kt < SEQ; kt += 32) {
        // prefetch next bias tile (global_prefetch_b8)
        if (kt + 32 < SEQ)
            __builtin_prefetch(bias + bias_base + (size_t)row0 * SEQ + kt + 32 + l16, 0, 0);
        // scores: S = (Q*scale) @ K^T, two 16-key tiles
        v8f sc[2];
        int km[2];
#pragma unroll
        for (int n = 0; n < 2; ++n) {
            const int key = kt + n * 16 + l16;
            const h16* kp = k + (size_t)(b * SEQ + key) * H_DIM + h * DKH;
            V16 bK0, bK1;
            {
                int kb0 = lh * 8;
                bK0.h[0] = ld8(kp + kb0);
                bK0.h[1] = ld8(kp + kb0 + 16);
                bK1.h[0] = ld8(kp + 32 + kb0);
                bK1.h[1] = ld8(kp + 32 + kb0 + 16);
            }
            v8f s = __builtin_amdgcn_wmma_f32_16x16x32_f16(
                false, aQ[0].v, false, bK0.v, (short)0, zf, false, false);
            s = __builtin_amdgcn_wmma_f32_16x16x32_f16(
                false, aQ[1].v, false, bK1.v, (short)0, s, false, false);
            sc[n] = s;
            km[n] = mask[b * SEQ + key];
        }
        // key-mask then bias (matches reference order)
#pragma unroll
        for (int n = 0; n < 2; ++n) {
            const int key = kt + n * 16 + l16;
#pragma unroll
            for (int r = 0; r < 8; ++r) {
                float s = sc[n][r];
                if (km[n] == 0) s = -1e9f;
                s += bias[bias_base + (size_t)(row0 + r) * SEQ + key];
                sc[n][r] = s;
            }
        }
        // online softmax; stage P into LDS for C->A re-layout
#pragma unroll
        for (int r = 0; r < 8; ++r) {
            float mx = fmaxf(sc[0][r], sc[1][r]);
#pragma unroll
            for (int off = 8; off; off >>= 1)
                mx = fmaxf(mx, __shfl_xor(mx, off, 32));
            const float mn = fmaxf(mrow[r], mx);
            corr[r] = __expf(mrow[r] - mn);
            mrow[r] = mn;
            const float p0 = __expf(sc[0][r] - mn);
            const float p1 = __expf(sc[1][r] - mn);
            float ps = p0 + p1;
#pragma unroll
            for (int off = 8; off; off >>= 1)
                ps += __shfl_xor(ps, off, 32);
            srow[r] = srow[r] * corr[r] + ps;
            const int prow = r + 8 * lh;
            Ps[prow * 48 + l16]      = (h16)p0;
            Ps[prow * 48 + 16 + l16] = (h16)p1;
        }
        __syncthreads();

        // P (16x32) as A fragment
        V16 aP;
        {
            const int kp0 = lh * 8;
            aP.h[0] = *(const v8h*)&Ps[l16 * 48 + kp0];
            aP.h[1] = *(const v8h*)&Ps[l16 * 48 + kp0 + 16];
        }
        // O += P @ V  (V^T layout -> contiguous key slices per dk column)
#pragma unroll
        for (int tdk = 0; tdk < 4; ++tdk) {
            const int dk = tdk * 16 + l16;
            const h16* vp = vt + (size_t)((b * NHEAD + h) * DKH + dk) * SEQ
                               + kt + lh * 8;
            V16 bV;
            bV.h[0] = ld8(vp);
            bV.h[1] = ld8(vp + 16);
#pragma unroll
            for (int r = 0; r < 8; ++r) oacc[tdk][r] *= corr[r];
            oacc[tdk] = __builtin_amdgcn_wmma_f32_16x16x32_f16(
                false, aP.v, false, bV.v, (short)0, oacc[tdk], false, false);
        }
        __syncthreads();
    }

    // normalize, apply query mask, store f16
#pragma unroll
    for (int r = 0; r < 8; ++r) {
        const int grow = qt * 16 + r + 8 * lh;
        const float inv = (srow[r] > 0.f) ? (1.0f / srow[r]) : 0.f;
        corr[r] = inv * (float)mask[b * SEQ + grow];
    }
#pragma unroll
    for (int tdk = 0; tdk < 4; ++tdk) {
#pragma unroll
        for (int r = 0; r < 8; ++r) {
            const int grow = qt * 16 + r + 8 * lh;
            o[(size_t)(b * SEQ + grow) * H_DIM + h * DKH + tdk * 16 + l16] =
                (h16)(oacc[tdk][r] * corr[r]);
        }
    }
}

// ---------------- orchestration -------------------------------------------
extern "C" void kernel_launch(void* const* d_in, const int* in_sizes, int n_in,
                              void* d_out, int out_size, void* d_ws, size_t ws_size,
                              hipStream_t stream) {
    const float* x    = (const float*)d_in[0];
    const float* ab   = (const float*)d_in[1];
    const int*   mask = (const int*)d_in[2];
    const float* ln1g = (const float*)d_in[3];
    const float* ln1b = (const float*)d_in[4];
    const float* Wq   = (const float*)d_in[5];
    const float* bq   = (const float*)d_in[6];
    const float* Wk   = (const float*)d_in[7];
    const float* bk   = (const float*)d_in[8];
    const float* Wv   = (const float*)d_in[9];
    const float* bv   = (const float*)d_in[10];
    const float* Wo   = (const float*)d_in[11];
    const float* bo   = (const float*)d_in[12];
    const float* ln2g = (const float*)d_in[13];
    const float* ln2b = (const float*)d_in[14];
    const float* W1   = (const float*)d_in[15];
    const float* b1   = (const float*)d_in[16];
    const float* W2   = (const float*)d_in[17];
    const float* b2   = (const float*)d_in[18];
    float* out = (float*)d_out;

    char* ws = (char*)d_ws;
    size_t off = 0;
    auto alloc = [&](size_t bytes) -> char* {
        char* p = ws + off;
        off = (off + bytes + 255) & ~(size_t)255;
        return p;
    };
    h16*   y1  = (h16*)alloc((size_t)ROWS * H_DIM * 2);
    h16*   wqT = (h16*)alloc((size_t)H_DIM * H_DIM * 2);
    h16*   wkT = (h16*)alloc((size_t)H_DIM * H_DIM * 2);
    h16*   wvT = (h16*)alloc((size_t)H_DIM * H_DIM * 2);
    h16*   woT = (h16*)alloc((size_t)H_DIM * H_DIM * 2);
    h16*   w1T = (h16*)alloc((size_t)H_DIM * FFN_D * 2);
    h16*   w2T = (h16*)alloc((size_t)FFN_D * H_DIM * 2);
    h16*   qf  = (h16*)alloc((size_t)ROWS * H_DIM * 2);
    h16*   kf  = (h16*)alloc((size_t)ROWS * H_DIM * 2);
    h16*   vf  = (h16*)alloc((size_t)ROWS * H_DIM * 2);
    h16*   vtf = (h16*)alloc((size_t)ROWS * H_DIM * 2);
    h16*   of  = (h16*)alloc((size_t)ROWS * H_DIM * 2);
    float* xo  = (float*)alloc((size_t)ROWS * H_DIM * 4);
    h16*   y2  = (h16*)alloc((size_t)ROWS * H_DIM * 2);
    h16*   h1  = (h16*)alloc((size_t)ROWS * FFN_D * 2);

    // 1. weights -> f16, transposed [N][K]
    {
        int blk1 = (H_DIM * H_DIM + 255) / 256;
        cvt_transpose_kernel<<<blk1, 256, 0, stream>>>(Wq, wqT, H_DIM, H_DIM);
        cvt_transpose_kernel<<<blk1, 256, 0, stream>>>(Wk, wkT, H_DIM, H_DIM);
        cvt_transpose_kernel<<<blk1, 256, 0, stream>>>(Wv, wvT, H_DIM, H_DIM);
        cvt_transpose_kernel<<<blk1, 256, 0, stream>>>(Wo, woT, H_DIM, H_DIM);
        int blk2 = (H_DIM * FFN_D + 255) / 256;
        cvt_transpose_kernel<<<blk2, 256, 0, stream>>>(W1, w1T, H_DIM, FFN_D);
        cvt_transpose_kernel<<<blk2, 256, 0, stream>>>(W2, w2T, FFN_D, H_DIM);
    }
    // 2. LN1
    ln_f16_kernel<<<ROWS, 256, 0, stream>>>(x, ln1g, ln1b, y1);
    // 3. QKV projections
    {
        dim3 g(H_DIM / 128, ROWS / 128);
        gemm_f16_kernel<false, false, false><<<g, 256, 0, stream>>>(
            y1, wqT, bq, nullptr, nullptr, qf, ROWS, H_DIM, H_DIM);
        gemm_f16_kernel<false, false, false><<<g, 256, 0, stream>>>(
            y1, wkT, bk, nullptr, nullptr, kf, ROWS, H_DIM, H_DIM);
        gemm_f16_kernel<false, false, false><<<g, 256, 0, stream>>>(
            y1, wvT, bv, nullptr, nullptr, vf, ROWS, H_DIM, H_DIM);
    }
    // 4. V transpose per head
    transpose_v_kernel<<<(ROWS * H_DIM + 255) / 256, 256, 0, stream>>>(vf, vtf);
    // 5. attention
    {
        dim3 g(SEQ / 16, NHEAD, BATCH);
        attn_kernel<<<g, 32, 0, stream>>>(qf, kf, vtf, ab, mask, of);
    }
    // 6. output projection + residual (fp32)
    {
        dim3 g(H_DIM / 128, ROWS / 128);
        gemm_f16_kernel<false, true, true><<<g, 256, 0, stream>>>(
            of, woT, bo, x, xo, nullptr, ROWS, H_DIM, H_DIM);
    }
    // 7. LN2
    ln_f16_kernel<<<ROWS, 256, 0, stream>>>(xo, ln2g, ln2b, y2);
    // 8. FFN1 + exact GELU (f16 out)
    {
        dim3 g(FFN_D / 128, ROWS / 128);
        gemm_f16_kernel<true, false, false><<<g, 256, 0, stream>>>(
            y2, w1T, b1, nullptr, nullptr, h1, ROWS, FFN_D, H_DIM);
    }
    // 9. FFN2 + residual -> d_out (fp32)
    {
        dim3 g(H_DIM / 128, ROWS / 128);
        gemm_f16_kernel<false, true, true><<<g, 256, 0, stream>>>(
            h1, w2T, b2, xo, out, nullptr, ROWS, H_DIM, FFN_D);
    }
    (void)in_sizes; (void)n_in; (void)out_size; (void)ws_size;
}
